// TreeAttentionLayer_69509750718660
// MI455X (gfx1250) — compile-verified, matrix operand order checked by reading, not run on hardware
//
#include <hip/hip_runtime.h>
#include <cstdint>
#include <cstddef>

// ---------------- problem constants ----------------
#define B_    16
#define S_    1024
#define D_    256
#define H_    8
#define DH_   32
#define DFF_  1024
#define PDIM_ 64
#define NTOK  (B_ * S_)      // 16384
#define E_    262144
#define EN_   (E_ + NTOK)    // 278528
#define EPSF  1e-5f

typedef __bf16 bf16;
typedef __attribute__((ext_vector_type(16))) __bf16 v16bf;
typedef __attribute__((ext_vector_type(8)))  float  v8f;

// ---------------- CDNA5 async copy to LDS (ASYNCcnt-tracked) ----------------
__device__ inline unsigned lds_off_u32(const void* p) {
  // LDS aperture: generic addr[31:0] is the LDS byte offset (ISA 10.2)
  return (unsigned)(uintptr_t)p;
}
__device__ inline void async_copy_b128(unsigned lds_off, const void* gbase, unsigned gbyte_off) {
  asm volatile("global_load_async_to_lds_b128 %0, %1, %2"
               :: "v"(lds_off), "v"(gbyte_off), "s"(gbase) : "memory");
}
__device__ inline void async_copy_b64(unsigned lds_off, const void* gbase, unsigned gbyte_off) {
  asm volatile("global_load_async_to_lds_b64 %0, %1, %2"
               :: "v"(lds_off), "v"(gbyte_off), "s"(gbase) : "memory");
}
__device__ inline void wait_async0() {
  asm volatile("s_wait_asynccnt 0" ::: "memory");
}

// ---------------- WMMA helpers ----------------
__device__ inline v8f wmma_bf16(v16bf a, v16bf b, v8f c) {
  return __builtin_amdgcn_wmma_f32_16x16x32_bf16(false, a, false, b, (short)0, c, false, false);
}

// Row-fragment loader (works for both A and [n][k]-stored B, LDS or global):
// lane L -> row (L&15) of the 16-row tile; element e -> K = e + (e&8) + 8*(L>>4).
// K-pairs (2i,2i+1) are contiguous -> 8x 32-bit loads.
__device__ inline v16bf load_frag_row(const bf16* base, int ldr, int lane) {
  union { v16bf v; unsigned u[8]; } r;
  const bf16* p = base + (size_t)(lane & 15) * ldr;
  int hk = (lane >> 4) * 8;
#pragma unroll
  for (int i = 0; i < 8; ++i) {
    int k = 2 * i + ((2 * i) & 8) + hk;  // 0,2,4,6,16,18,20,22 (+8 on hi half)
    r.u[i] = *(const unsigned*)(p + k);
  }
  return r.v;
}

__device__ inline float wred_sum(float s) {
  s += __shfl_xor(s, 16); s += __shfl_xor(s, 8); s += __shfl_xor(s, 4);
  s += __shfl_xor(s, 2);  s += __shfl_xor(s, 1);
  return s;
}

__device__ inline void atomicMaxF(float* addr, float val) {
  int* ia = (int*)addr;
  int old = __float_as_int(*addr);
  while (__int_as_float(old) < val) {
    int prev = atomicCAS(ia, old, __float_as_int(val));
    if (prev == old) break;
    old = prev;
  }
}

__device__ inline float gelu_exact(float v) {
  return 0.5f * v * (1.0f + erff(v * 0.70710678118654752f));
}

// ---------------- weight pre-pack: f32 -> bf16 (layout [N][K] == original [out][in]) ----------
__global__ __launch_bounds__(256) void pack_w_kernel(const float* __restrict__ W,
                                                     bf16* __restrict__ out, int total) {
  int idx = blockIdx.x * 256 + threadIdx.x;
  if (idx >= total) return;
  out[idx] = (bf16)W[idx];
}

// ---------------- AdaLN gamma/beta ----------------
__global__ __launch_bounds__(256) void ada_params_kernel(
    const float* __restrict__ prop,
    const float* __restrict__ g1w, const float* __restrict__ g1b,
    const float* __restrict__ b1w, const float* __restrict__ b1b,
    const float* __restrict__ g2w, const float* __restrict__ g2b,
    const float* __restrict__ b2w, const float* __restrict__ b2b,
    float* __restrict__ og1, float* __restrict__ ob1,
    float* __restrict__ og2, float* __restrict__ ob2) {
  int b = blockIdx.x, d = threadIdx.x;
  float s1 = g1b[d], s2 = b1b[d], s3 = g2b[d], s4 = b2b[d];
#pragma unroll 4
  for (int p = 0; p < PDIM_; ++p) {
    float pc = prop[b * PDIM_ + p];
    s1 += pc * g1w[d * PDIM_ + p];
    s2 += pc * b1w[d * PDIM_ + p];
    s3 += pc * g2w[d * PDIM_ + p];
    s4 += pc * b2w[d * PDIM_ + p];
  }
  og1[b * D_ + d] = s1; ob1[b * D_ + d] = s2;
  og2[b * D_ + d] = s3; ob2[b * D_ + d] = s4;
}

// ---------------- fused AdaLN1 + LN1 -> xn1 (bf16) ----------------
__global__ __launch_bounds__(256) void adaln_pre_kernel(
    const float* __restrict__ x, const float* __restrict__ g1, const float* __restrict__ be1,
    const float* __restrict__ ln1w, const float* __restrict__ ln1b,
    bf16* __restrict__ xn1) {
  int row = blockIdx.x * 8 + (threadIdx.x >> 5);
  int lane = threadIdx.x & 31;
  int b = row >> 10;
  float v[8];
#pragma unroll
  for (int i = 0; i < 8; ++i) v[i] = x[(size_t)row * D_ + i * 32 + lane];
  float s = 0.f;
#pragma unroll
  for (int i = 0; i < 8; ++i) s += v[i];
  float mu = wred_sum(s) * (1.0f / D_);
  s = 0.f;
#pragma unroll
  for (int i = 0; i < 8; ++i) { float t = v[i] - mu; s += t * t; }
  float rs = rsqrtf(wred_sum(s) * (1.0f / D_) + EPSF);
  float xa[8];
#pragma unroll
  for (int i = 0; i < 8; ++i) {
    int d = i * 32 + lane;
    xa[i] = g1[b * D_ + d] * ((v[i] - mu) * rs) + be1[b * D_ + d];
  }
  s = 0.f;
#pragma unroll
  for (int i = 0; i < 8; ++i) s += xa[i];
  float mu2 = wred_sum(s) * (1.0f / D_);
  s = 0.f;
#pragma unroll
  for (int i = 0; i < 8; ++i) { float t = xa[i] - mu2; s += t * t; }
  float rs2 = rsqrtf(wred_sum(s) * (1.0f / D_) + EPSF);
#pragma unroll
  for (int i = 0; i < 8; ++i) {
    int d = i * 32 + lane;
    xn1[(size_t)row * D_ + d] = (bf16)((xa[i] - mu2) * rs2 * ln1w[d] + ln1b[d]);
  }
}

// ---------------- WMMA GEMM: C[M,N] = A[M,K](bf16) @ W[N,K](bf16)^T ----------------
// Epilogues: 0 raw->f32 ; 1 +bias->bf16 ; 2 +bias,GELU->bf16 ;
//            3 +bias,+res->f32&bf16 ; 4 +bias,+res->f32
template <int EPI>
__global__ __launch_bounds__(256) void gemm_bf16_kernel(
    const bf16* __restrict__ A, const bf16* __restrict__ Wb,
    const float* __restrict__ bias, const float* __restrict__ residual,
    float* __restrict__ Cf, bf16* __restrict__ Cb,
    int M, int Ncols, int K) {
  __shared__ alignas(16) bf16 As[128][40];   // 128x32 A tile, 80B rows (16B aligned)
  __shared__ alignas(16) bf16 Bs[64][40];    // 64x32 W tile ([n][k]), 80B rows

  int t = threadIdx.x;
  int lane = t & 31, wv = t >> 5;
  int wm = wv & 3, wn = wv >> 2;             // 4x2 wave grid, 32x32 per wave
  int hi = lane >> 4, ln = lane & 15;
  int mBase = blockIdx.y * 128;
  int nBase = blockIdx.x * 64;

  int ar = t >> 1, ac = (t & 1) * 16;        // A staging: 16 bf16 per thread (2x b128)
  int bn = t >> 2, bk = (t & 3) * 8;         // W staging: 8 bf16 per thread (1x b128)
  unsigned ldsA0 = lds_off_u32(&As[ar][ac]);
  unsigned ldsA1 = ldsA0 + 16;
  unsigned ldsB  = lds_off_u32(&Bs[bn][bk]);

  v8f acc[2][2] = {};
  for (int k0 = 0; k0 < K; k0 += 32) {
    unsigned aoff = (unsigned)(((size_t)(mBase + ar) * K + k0 + ac) * 2);
    unsigned boff = (unsigned)(((size_t)(nBase + bn) * K + k0 + bk) * 2);
    async_copy_b128(ldsA0, A, aoff);
    async_copy_b128(ldsA1, A, aoff + 16);
    async_copy_b128(ldsB, Wb, boff);
    wait_async0();
    __syncthreads();

    v16bf a0 = load_frag_row(&As[wm * 32][0], 40, lane);
    v16bf a1 = load_frag_row(&As[wm * 32 + 16][0], 40, lane);
    v16bf b0 = load_frag_row(&Bs[wn * 32][0], 40, lane);
    v16bf b1 = load_frag_row(&Bs[wn * 32 + 16][0], 40, lane);
    acc[0][0] = wmma_bf16(a0, b0, acc[0][0]);
    acc[0][1] = wmma_bf16(a0, b1, acc[0][1]);
    acc[1][0] = wmma_bf16(a1, b0, acc[1][0]);
    acc[1][1] = wmma_bf16(a1, b1, acc[1][1]);
    __syncthreads();
  }

#pragma unroll
  for (int i = 0; i < 2; ++i)
#pragma unroll
    for (int j = 0; j < 2; ++j) {
#pragma unroll
      for (int e = 0; e < 8; ++e) {
        int r = mBase + wm * 32 + i * 16 + e + 8 * hi;
        int c = nBase + wn * 32 + j * 16 + ln;
        float v = acc[i][j][e];
        if (EPI >= 1) v += bias[c];
        if (EPI == 2) v = gelu_exact(v);
        if (EPI == 3 || EPI == 4) v += residual[(size_t)r * Ncols + c];
        if (EPI == 0 || EPI == 3 || EPI == 4) Cf[(size_t)r * Ncols + c] = v;
        if (EPI == 1 || EPI == 2 || EPI == 3) Cb[(size_t)r * Ncols + c] = (bf16)v;
      }
    }
}

// ---------------- flash attention (bf16 WMMA, S^T trick, LDS-staged K/V) ----------------
// qkv: [NTOK, 768] bf16 (Q|K|V, head-major 8x32);  ao: [NTOK, 256] bf16
// All 8 waves of a block share (b,h); each wave owns a 16-row q tile.
__global__ __launch_bounds__(256) void attention_kernel(const bf16* __restrict__ qkv,
                                                        bf16* __restrict__ ao) {
  const float CS = 0.17677669529663689f * 1.4426950408889634f;  // 1/sqrt(32) * log2(e)
  __shared__ alignas(16) bf16 Ks[32][40];   // K block, [k][d]
  __shared__ alignas(16) bf16 Vt[32][40];   // V block transposed, [d][k]

  int t = threadIdx.x;
  int lane = t & 31;
  int wid = blockIdx.x * 8 + (t >> 5);
  int qt = wid & 63;
  int bh = wid >> 6;                         // uniform within a block
  int b = bh >> 3, h = bh & 7;
  int bS = b * S_;
  int hi = lane >> 4, ln = lane & 15;
  int hk = hi * 8;

  // staging roles (block-wide): K via async b64, V transposed manually
  int tk = t >> 3, td = (t & 7) * 4;         // 32 rows x 8 chunks of 4 bf16
  unsigned ldsK = lds_off_u32(&Ks[tk][td]);

  // Q fragment (B-type for S^T = K x Q^T): lane = q column, elements = DH
  v16bf qf = load_frag_row(qkv + (size_t)(bS + qt * 16) * 768 + h * 32, 768, lane);

  v8f o0 = {}, o1 = {};
  float m_run = -1e30f, l_run = 0.f;

  for (int kb = 0; kb < S_ / 32; ++kb) {
    int kr = kb * 32;
    // stage K rows [kr..kr+31] of head h
    unsigned koff = (unsigned)(((size_t)(bS + kr + tk) * 768 + 256 + h * 32 + td) * 2);
    async_copy_b64(ldsK, qkv, koff);
    // stage V transposed
    const bf16* vr = qkv + (size_t)(bS + kr + tk) * 768 + 512 + h * 32 + td;
#pragma unroll
    for (int j = 0; j < 4; ++j) Vt[td + j][tk] = vr[j];
    wait_async0();
    __syncthreads();

    v16bf ka0 = load_frag_row(&Ks[0][0], 40, lane);
    v16bf ka1 = load_frag_row(&Ks[16][0], 40, lane);
    v16bf vb0 = load_frag_row(&Vt[0][0], 40, lane);
    v16bf vb1 = load_frag_row(&Vt[16][0], 40, lane);
    v8f z = {};
    v8f s0 = wmma_bf16(ka0, qf, z);  // S^T tile: k = e + 8*hi, q = ln
    v8f s1 = wmma_bf16(ka1, qf, z);  // k = 16 + e + 8*hi

    // online softmax over k; row q = ln (lanes L, L^16 share a row)
    float bm = -1e30f;
#pragma unroll
    for (int e = 0; e < 8; ++e) bm = fmaxf(bm, fmaxf(s0[e], s1[e]));
    bm = fmaxf(bm, __shfl_xor(bm, 16));
    float mn = fmaxf(m_run, bm);
    float alpha = exp2f((m_run - mn) * CS);
    float rsum = 0.f;
    v16bf pf;  // C-layout of s0/s1 == A-layout for P@V (zero shuffles)
#pragma unroll
    for (int e = 0; e < 8; ++e) {
      float p0 = exp2f((s0[e] - mn) * CS);
      float p1 = exp2f((s1[e] - mn) * CS);
      rsum += p0 + p1;
      pf[e] = (bf16)p0;
      pf[e + 8] = (bf16)p1;
    }
    rsum += __shfl_xor(rsum, 16);
    l_run = l_run * alpha + rsum;
    m_run = mn;
#pragma unroll
    for (int e = 0; e < 8; ++e) {  // rescale O (element space: row q = e + 8*hi)
      float ae = __shfl(alpha, e + hk);
      o0[e] *= ae;
      o1[e] *= ae;
    }
    o0 = wmma_bf16(pf, vb0, o0);
    o1 = wmma_bf16(pf, vb1, o1);
    __syncthreads();
  }

#pragma unroll
  for (int e = 0; e < 8; ++e) {
    float le = __shfl(l_run, e + hk);
    float inv = 1.0f / le;
    int tok = bS + qt * 16 + e + 8 * hi;
    ao[(size_t)tok * D_ + h * 32 + ln] = (bf16)(o0[e] * inv);
    ao[(size_t)tok * D_ + h * 32 + 16 + ln] = (bf16)(o1[e] * inv);
  }
}

// ---------------- GAT pieces ----------------
__global__ __launch_bounds__(256) void gat_scores_kernel(const float* __restrict__ h,
                                                         const float* __restrict__ att_src,
                                                         const float* __restrict__ att_dst,
                                                         float* __restrict__ a_src,
                                                         float* __restrict__ a_dst) {
  int row = blockIdx.x * 8 + (threadIdx.x >> 5);
  int lane = threadIdx.x & 31;
  float s1 = 0.f, s2 = 0.f;
#pragma unroll
  for (int i = 0; i < 8; ++i) {
    int d = i * 32 + lane;
    float hv = h[(size_t)row * D_ + d];
    s1 += hv * att_src[d];
    s2 += hv * att_dst[d];
  }
  s1 = wred_sum(s1);
  s2 = wred_sum(s2);
  if (lane == 0) { a_src[row] = s1; a_dst[row] = s2; }
}

__global__ __launch_bounds__(256) void gat_init_kernel(float* __restrict__ agg,
                                                       float* __restrict__ mmax,
                                                       float* __restrict__ zsum) {
  int idx = blockIdx.x * 256 + threadIdx.x;
  agg[idx] = 0.f;  // grid sized exactly NTOK*D_
  if (idx < NTOK) { mmax[idx] = -1e30f; zsum[idx] = 0.f; }
}

__global__ __launch_bounds__(256) void gat_edge1_kernel(const int* __restrict__ ei,
                                                        const float* __restrict__ a_src,
                                                        const float* __restrict__ a_dst,
                                                        float* __restrict__ eraw,
                                                        float* __restrict__ mmax) {
  int i = blockIdx.x * 256 + threadIdx.x;
  if (i >= EN_) return;
  int s, d;
  if (i < E_) { s = ei[i]; d = ei[E_ + i]; } else { s = i - E_; d = s; }
  float raw = a_src[s] + a_dst[d];
  float e = raw > 0.f ? raw : 0.2f * raw;  // leaky_relu 0.2
  eraw[i] = e;
  atomicMaxF(&mmax[d], e);
}

__global__ __launch_bounds__(256) void gat_edge2_kernel(const int* __restrict__ ei,
                                                        const float* __restrict__ eraw,
                                                        const float* __restrict__ mmax,
                                                        float* __restrict__ pbuf,
                                                        float* __restrict__ zsum) {
  int i = blockIdx.x * 256 + threadIdx.x;
  if (i >= EN_) return;
  int d = (i < E_) ? ei[E_ + i] : (i - E_);
  float p = __expf(eraw[i] - mmax[d]);
  pbuf[i] = p;
  atomicAdd(&zsum[d], p);
}

__global__ __launch_bounds__(256) void gat_edge3_kernel(const int* __restrict__ ei,
                                                        const float* __restrict__ pbuf,
                                                        const float* __restrict__ zsum,
                                                        const float* __restrict__ h,
                                                        float* __restrict__ agg) {
  int tid = blockIdx.x * 256 + threadIdx.x;  // (EN_*32) threads, 8 features each
  int i = tid >> 5;
  int c0 = (tid & 31) * 8;
  int s, d;
  if (i < E_) { s = ei[i]; d = ei[E_ + i]; } else { s = i - E_; d = s; }
  float alpha = pbuf[i] / zsum[d];
  const float* hs = h + (size_t)s * D_ + c0;
  float* ag = agg + (size_t)d * D_ + c0;
#pragma unroll
  for (int j = 0; j < 8; ++j) atomicAdd(&ag[j], alpha * hs[j]);
}

// ---------------- fused (x1+agg+bias) -> gLN -> AdaLN2 -> LN2 ----------------
__global__ __launch_bounds__(256) void post_gat_ln_kernel(
    const float* __restrict__ x1, const float* __restrict__ agg,
    const float* __restrict__ gat_bias,
    const float* __restrict__ glnw, const float* __restrict__ glnb,
    const float* __restrict__ g2, const float* __restrict__ be2,
    const float* __restrict__ ln2w, const float* __restrict__ ln2b,
    float* __restrict__ x3, bf16* __restrict__ xn2) {
  int row = blockIdx.x * 8 + (threadIdx.x >> 5);
  int lane = threadIdx.x & 31;
  int b = row >> 10;
  float v[8];
#pragma unroll
  for (int i = 0; i < 8; ++i) {
    int d = i * 32 + lane;
    v[i] = x1[(size_t)row * D_ + d] + agg[(size_t)row * D_ + d] + gat_bias[d];
  }
  float s = 0.f;
#pragma unroll
  for (int i = 0; i < 8; ++i) s += v[i];
  float mu = wred_sum(s) * (1.0f / D_);
  s = 0.f;
#pragma unroll
  for (int i = 0; i < 8; ++i) { float t = v[i] - mu; s += t * t; }
  float rs = rsqrtf(wred_sum(s) * (1.0f / D_) + EPSF);
  float x3v[8];
#pragma unroll
  for (int i = 0; i < 8; ++i) {
    int d = i * 32 + lane;
    x3v[i] = (v[i] - mu) * rs * glnw[d] + glnb[d];
    x3[(size_t)row * D_ + d] = x3v[i];
  }
  s = 0.f;
#pragma unroll
  for (int i = 0; i < 8; ++i) s += x3v[i];
  float mu2 = wred_sum(s) * (1.0f / D_);
  s = 0.f;
#pragma unroll
  for (int i = 0; i < 8; ++i) { float t = x3v[i] - mu2; s += t * t; }
  float rs2 = rsqrtf(wred_sum(s) * (1.0f / D_) + EPSF);
  float xa[8];
#pragma unroll
  for (int i = 0; i < 8; ++i) {
    int d = i * 32 + lane;
    xa[i] = g2[b * D_ + d] * ((x3v[i] - mu2) * rs2) + be2[b * D_ + d];
  }
  s = 0.f;
#pragma unroll
  for (int i = 0; i < 8; ++i) s += xa[i];
  float mu3 = wred_sum(s) * (1.0f / D_);
  s = 0.f;
#pragma unroll
  for (int i = 0; i < 8; ++i) { float t = xa[i] - mu3; s += t * t; }
  float rs3 = rsqrtf(wred_sum(s) * (1.0f / D_) + EPSF);
#pragma unroll
  for (int i = 0; i < 8; ++i) {
    int d = i * 32 + lane;
    xn2[(size_t)row * D_ + d] = (bf16)((xa[i] - mu3) * rs3 * ln2w[d] + ln2b[d]);
  }
}

// ---------------- host launch ----------------
extern "C" void kernel_launch(void* const* d_in, const int* in_sizes, int n_in,
                              void* d_out, int out_size, void* d_ws, size_t ws_size,
                              hipStream_t stream) {
  const float* x         = (const float*)d_in[0];
  const float* prop      = (const float*)d_in[1];
  const int*   edge      = (const int*)d_in[2];
  const float* ada1_gw   = (const float*)d_in[3];
  const float* ada1_gb   = (const float*)d_in[4];
  const float* ada1_bw   = (const float*)d_in[5];
  const float* ada1_bb   = (const float*)d_in[6];
  const float* ada2_gw   = (const float*)d_in[7];
  const float* ada2_gb   = (const float*)d_in[8];
  const float* ada2_bw   = (const float*)d_in[9];
  const float* ada2_bb   = (const float*)d_in[10];
  const float* ln1_w     = (const float*)d_in[11];
  const float* ln1_b     = (const float*)d_in[12];
  const float* ln2_w     = (const float*)d_in[13];
  const float* ln2_b     = (const float*)d_in[14];
  const float* gln_w     = (const float*)d_in[15];
  const float* gln_b     = (const float*)d_in[16];
  const float* in_proj_w = (const float*)d_in[17];
  const float* in_proj_b = (const float*)d_in[18];
  const float* out_proj_w= (const float*)d_in[19];
  const float* out_proj_b= (const float*)d_in[20];
  const float* ffn_w1    = (const float*)d_in[21];
  const float* ffn_b1    = (const float*)d_in[22];
  const float* ffn_w2    = (const float*)d_in[23];
  const float* ffn_b2    = (const float*)d_in[24];
  const float* gat_w     = (const float*)d_in[25];
  const float* gat_att_s = (const float*)d_in[26];
  const float* gat_att_d = (const float*)d_in[27];
  const float* gat_bias  = (const float*)d_in[28];
  float* out = (float*)d_out;

  char* ws = (char*)d_ws;
  size_t off = 0;
  auto carve = [&](size_t bytes) -> char* {
    off = (off + 255) & ~(size_t)255;
    char* p = ws + off;
    off += bytes;
    return p;
  };
  float* g1   = (float*)carve(B_ * D_ * 4);
  float* be1  = (float*)carve(B_ * D_ * 4);
  float* g2   = (float*)carve(B_ * D_ * 4);
  float* be2  = (float*)carve(B_ * D_ * 4);
  bf16* Wqkv  = (bf16*)carve((size_t)3 * D_ * D_ * 2);   // [768][256]
  bf16* Wo    = (bf16*)carve((size_t)D_ * D_ * 2);       // [256][256]
  bf16* Wg    = (bf16*)carve((size_t)D_ * D_ * 2);
  bf16* W1    = (bf16*)carve((size_t)DFF_ * D_ * 2);     // [1024][256]
  bf16* W2    = (bf16*)carve((size_t)D_ * DFF_ * 2);     // [256][1024]
  bf16* xn1   = (bf16*)carve((size_t)NTOK * D_ * 2);
  bf16* qkv   = (bf16*)carve((size_t)NTOK * 3 * D_ * 2);
  bf16* ao    = (bf16*)carve((size_t)NTOK * D_ * 2);
  float* x1f  = (float*)carve((size_t)NTOK * D_ * 4);
  bf16* x1b   = (bf16*)carve((size_t)NTOK * D_ * 2);
  float* hf   = (float*)carve((size_t)NTOK * D_ * 4);
  float* asrc = (float*)carve((size_t)NTOK * 4);
  float* adst = (float*)carve((size_t)NTOK * 4);
  float* eraw = (float*)carve((size_t)EN_ * 4);
  float* pbuf = (float*)carve((size_t)EN_ * 4);
  float* mmax = (float*)carve((size_t)NTOK * 4);
  float* zsum = (float*)carve((size_t)NTOK * 4);
  float* agg  = (float*)carve((size_t)NTOK * D_ * 4);
  float* x3f  = (float*)carve((size_t)NTOK * D_ * 4);
  bf16* xn2   = (bf16*)carve((size_t)NTOK * D_ * 2);
  bf16* ff1   = (bf16*)carve((size_t)NTOK * DFF_ * 2);

  // 1) weights -> bf16 (same [out][in] layout == [N][K])
  pack_w_kernel<<<(3 * D_ * D_ + 255) / 256, 256, 0, stream>>>(in_proj_w, Wqkv, 3 * D_ * D_);
  pack_w_kernel<<<(D_ * D_ + 255) / 256, 256, 0, stream>>>(out_proj_w, Wo, D_ * D_);
  pack_w_kernel<<<(D_ * D_ + 255) / 256, 256, 0, stream>>>(gat_w, Wg, D_ * D_);
  pack_w_kernel<<<(DFF_ * D_ + 255) / 256, 256, 0, stream>>>(ffn_w1, W1, DFF_ * D_);
  pack_w_kernel<<<(D_ * DFF_ + 255) / 256, 256, 0, stream>>>(ffn_w2, W2, D_ * DFF_);

  // 2) AdaLN conditioning params
  ada_params_kernel<<<B_, 256, 0, stream>>>(prop, ada1_gw, ada1_gb, ada1_bw, ada1_bb,
                                            ada2_gw, ada2_gb, ada2_bw, ada2_bb,
                                            g1, be1, g2, be2);

  // 3) fused AdaLN1 + LN1
  adaln_pre_kernel<<<NTOK / 8, 256, 0, stream>>>(x, g1, be1, ln1_w, ln1_b, xn1);

  // 4) QKV projection
  gemm_bf16_kernel<1><<<dim3(3 * D_ / 64, NTOK / 128), 256, 0, stream>>>(
      xn1, Wqkv, in_proj_b, nullptr, nullptr, qkv, NTOK, 3 * D_, D_);

  // 5) attention
  attention_kernel<<<(B_ * H_ * (S_ / 16)) / 8, 256, 0, stream>>>(qkv, ao);

  // 6) out_proj + residual(x)
  gemm_bf16_kernel<3><<<dim3(D_ / 64, NTOK / 128), 256, 0, stream>>>(
      ao, Wo, out_proj_b, x, x1f, x1b, NTOK, D_, D_);

  // 7) GAT h = x1 @ gat_w^T
  gemm_bf16_kernel<0><<<dim3(D_ / 64, NTOK / 128), 256, 0, stream>>>(
      x1b, Wg, nullptr, nullptr, hf, nullptr, NTOK, D_, D_);

  // 8) segment softmax over edges
  gat_scores_kernel<<<NTOK / 8, 256, 0, stream>>>(hf, gat_att_s, gat_att_d, asrc, adst);
  gat_init_kernel<<<(NTOK * D_) / 256, 256, 0, stream>>>(agg, mmax, zsum);
  gat_edge1_kernel<<<(EN_ + 255) / 256, 256, 0, stream>>>(edge, asrc, adst, eraw, mmax);
  gat_edge2_kernel<<<(EN_ + 255) / 256, 256, 0, stream>>>(edge, eraw, mmax, pbuf, zsum);
  gat_edge3_kernel<<<((size_t)EN_ * 32) / 256, 256, 0, stream>>>(edge, pbuf, zsum, hf, agg);

  // 9) fused residual + gLN + AdaLN2 + LN2
  post_gat_ln_kernel<<<NTOK / 8, 256, 0, stream>>>(x1f, agg, gat_bias, gln_w, gln_b,
                                                   g2, be2, ln2_w, ln2_b, x3f, xn2);

  // 10) FFN1 + GELU
  gemm_bf16_kernel<2><<<dim3(DFF_ / 64, NTOK / 128), 256, 0, stream>>>(
      xn2, W1, ffn_b1, nullptr, nullptr, ff1, NTOK, DFF_, D_);

  // 11) FFN2 + residual(x3) -> d_out
  gemm_bf16_kernel<4><<<dim3(D_ / 64, NTOK / 128), 256, 0, stream>>>(
      ff1, W2, ffn_b2, x3f, out, nullptr, NTOK, D_, DFF_);
}
